// PGN_39797166965178
// MI455X (gfx1250) — matrix operand order repaired and checked by reading, not compiled, not run on hardware
//
#include <hip/hip_runtime.h>

// Problem constants (match reference)
#define B_   2
#define N_   512
#define DN   128
#define MID_ 128
#define OUT_ 256
#define ZIN  256
#define LAY  2
#define SEGS 16
#define SEGLEN (N_/SEGS)   // 32 senders per workgroup

typedef __attribute__((ext_vector_type(16))) _Float16 v16h;
typedef __attribute__((ext_vector_type(8)))  _Float16 v8h;
typedef __attribute__((ext_vector_type(8)))  float    v8f;

// order-preserving float -> uint mapping (for atomic u32 max on floats)
__device__ __forceinline__ unsigned fmap(float f) {
    unsigned u = __float_as_uint(f);
    return (u & 0x80000000u) ? ~u : (u | 0x80000000u);
}
__device__ __forceinline__ float funmap(unsigned u) {
    unsigned b = (u & 0x80000000u) ? (u & 0x7fffffffu) : ~u;
    return __uint_as_float(b);
}

// ---------------------------------------------------------------------------
// K0: z0 = concat(node_fts, hidden)
// ---------------------------------------------------------------------------
__global__ void k_init_z(const float* __restrict__ node,
                         const float* __restrict__ hid,
                         float* __restrict__ z0) {
    int t = blockIdx.x * blockDim.x + threadIdx.x;   // B*N*DN threads
    if (t >= B_ * N_ * DN) return;
    int bn = t >> 7, d = t & 127;
    z0[(size_t)bn * ZIN + d]        = node[t];
    z0[(size_t)bn * ZIN + 128 + d]  = hid[t];
}

// ---------------------------------------------------------------------------
// K1: msg1 = z@m1_w + m1_b ; msg2 = z@m2_w + m2_b ;
//     msgg = graph@mg_w + mg_b + me_b (constants pulled out of the max)
//     Also resets the atomic-max accumulator.
// grid = B*N blocks x 128 threads
// ---------------------------------------------------------------------------
__global__ void k_msgs(const float* __restrict__ z,  const float* __restrict__ gf,
                       const float* __restrict__ m1w, const float* __restrict__ m1b,
                       const float* __restrict__ m2w, const float* __restrict__ m2b,
                       const float* __restrict__ mgw, const float* __restrict__ mgb,
                       const float* __restrict__ meb,
                       float* __restrict__ msg1, float* __restrict__ msg2,
                       float* __restrict__ msgg, unsigned* __restrict__ redu) {
    __shared__ float zs[ZIN];
    int bn = blockIdx.x, m = threadIdx.x;            // 128 threads
    zs[m]       = z[(size_t)bn * ZIN + m];
    zs[m + 128] = z[(size_t)bn * ZIN + 128 + m];
    redu[(size_t)bn * MID_ + m] = fmap(-1.0e6f);     // reset reduction buffer
    __syncthreads();
    float a1 = m1b[m], a2 = m2b[m];
#pragma unroll 4
    for (int k = 0; k < ZIN; ++k) {
        float zv = zs[k];
        a1 += zv * m1w[k * MID_ + m];
        a2 += zv * m2w[k * MID_ + m];
    }
    msg1[(size_t)bn * MID_ + m] = a1;
    msg2[(size_t)bn * MID_ + m] = a2;
    int n = bn & (N_ - 1), b = bn >> 9;              // N_ == 512
    if (n == 0) {
        float ag = mgb[m] + meb[m];
        for (int k = 0; k < DN; ++k) ag += gf[b * DN + k] * mgw[k * MID_ + m];
        msgg[b * MID_ + m] = ag;
    }
}

// ---------------------------------------------------------------------------
// K2: the big one.  For each (b, receiver-tile j0..j0+15, sender-segment):
//   inner[b,j,m] = max over i in segment, adj[b,i,j]!=0, of
//                  ( edge[b,i,j,:] @ me_w[:,m]  +  msg2[b,i,m] )
// merged across segments with atomic u32 max.
// Block = 256 threads (8 waves); wave w owns mid columns m0 = 16*w.
// A tile (16 receivers x K=32 chunk, f16) staged via LDS in WMMA A-fragment
// order; me_w B-fragments preloaded in registers; msg2 injected as WMMA C.
// ---------------------------------------------------------------------------
__global__ void __launch_bounds__(256) k_edge_max(
        const float* __restrict__ edge, const int* __restrict__ adj,
        const float* __restrict__ msg2, const float* __restrict__ mew,
        unsigned* __restrict__ redu) {
    __shared__ __align__(32) _Float16 aLds[2][4 * 32 * 16];   // 2 x 4KB

    const int tid  = threadIdx.x;
    const int lane = tid & 31, wave = tid >> 5;
    const int g    = lane >> 4, nn = lane & 15;
    const int m0   = wave * 16;
    const int seg  = blockIdx.x, jt = blockIdx.y, b = blockIdx.z;
    const int j0   = jt * 16, i0 = seg * SEGLEN;

    // Preload B fragments (me_w, f32 -> f16). B layout: lane group g holds
    // K = ch*32 + g*16 + e ; column N = nn.
    v16h bf[4];
#pragma unroll
    for (int cb = 0; cb < 4; ++cb) {
#pragma unroll
        for (int e = 0; e < 16; ++e)
            bf[cb][e] = (_Float16)mew[(cb * 32 + g * 16 + e) * MID_ + m0 + nn];
    }

    float acc[8];
#pragma unroll
    for (int r = 0; r < 8; ++r) acc[r] = -1.0e6f;

    // Cooperative A-tile staging: thread -> (row M, 8 consecutive K), stored
    // pre-swizzled so fragment reads are contiguous b128s.
    const int Mrow = tid >> 4;            // 0..15 receiver row within tile
    const int col0 = (tid & 15) * 8;      // 0,8,...,120 feature offset
    const int chnk = col0 >> 5;
    const int kk0  = col0 & 31;
    const int sg   = (kk0 >> 3) & 1;
    const int hi   = kk0 >> 4;
    const int ldsIdx = ((chnk * 32) + ((hi << 4) | Mrow)) * 16 + sg * 8;

    float4 f0, f1;
    {
        const float* src = edge + ((((size_t)b * N_ + i0) * N_) + (j0 + Mrow)) * 128 + col0;
        f0 = *(const float4*)src;  f1 = *(const float4*)(src + 4);
    }

    int p = 0;
    for (int it = 0; it < SEGLEN; ++it) {
        const int i = i0 + it;
        {   // convert + scatter into LDS buffer p
            v8h h;
            h[0] = (_Float16)f0.x; h[1] = (_Float16)f0.y;
            h[2] = (_Float16)f0.z; h[3] = (_Float16)f0.w;
            h[4] = (_Float16)f1.x; h[5] = (_Float16)f1.y;
            h[6] = (_Float16)f1.z; h[7] = (_Float16)f1.w;
            *(v8h*)&aLds[p][ldsIdx] = h;
        }
        __syncthreads();   // one barrier/iter is safe with double buffering

        if (it + 1 < SEGLEN) {   // prefetch next sender's rows into registers
            const float* src = edge + ((((size_t)b * N_ + (i + 1)) * N_) + (j0 + Mrow)) * 128 + col0;
            f0 = *(const float4*)src;  f1 = *(const float4*)(src + 4);
        }

        // msg2[b,i,m] is constant per output column -> use as WMMA C operand
        const float m2v = msg2[((size_t)b * N_ + i) * MID_ + m0 + nn];
        const int* adjRow = adj + ((size_t)b * N_ + i) * N_ + j0;   // uniform -> s_load

        v8f c;
#pragma unroll
        for (int r = 0; r < 8; ++r) c[r] = m2v;
#pragma unroll
        for (int ch = 0; ch < 4; ++ch) {
            v16h a = *(const v16h*)&aLds[p][(ch * 32 + lane) * 16];
            c = __builtin_amdgcn_wmma_f32_16x16x32_f16(
                    false, a, false, bf[ch], (short)0, c, false, false);
        }
        // masked max over this sender; D row M=(g*8+r) <-> receiver j0+g*8+r.
        // Pure compare-and-select: v_cmp_gt_f32 needs no canonicalized
        // inputs, and v_cndmask is a bit-move, so no v_max v,v,v
        // canonicalize ops appear in the hot loop.
#pragma unroll
        for (int r = 0; r < 8; ++r) {
            int alo = adjRow[r], ahi = adjRow[r + 8];
            int av  = g ? ahi : alo;
            bool upd = (av != 0) & (c[r] > acc[r]);
            acc[r]   = upd ? c[r] : acc[r];
        }
        p ^= 1;
    }

    // merge segment-partial maxes
#pragma unroll
    for (int r = 0; r < 8; ++r) {
        int j = j0 + g * 8 + r;
        atomicMax(&redu[((size_t)b * N_ + j) * MID_ + m0 + nn], fmap(acc[r]));
    }
}

// ---------------------------------------------------------------------------
// K3: z_out = relu( z@o1_w + o1_b + (inner + msg1 + msgg)@o2_w + o2_b )
// grid = B*N blocks x 256 threads
// ---------------------------------------------------------------------------
__global__ void k_out(const float* __restrict__ z, const unsigned* __restrict__ redu,
                      const float* __restrict__ msg1, const float* __restrict__ msgg,
                      const float* __restrict__ o1w, const float* __restrict__ o1b,
                      const float* __restrict__ o2w, const float* __restrict__ o2b,
                      float* __restrict__ zout) {
    __shared__ float zs[ZIN];
    __shared__ float ms[MID_];
    int bn = blockIdx.x, t = threadIdx.x;  // 256 threads
    int b = bn >> 9;
    zs[t] = z[(size_t)bn * ZIN + t];
    if (t < MID_)
        ms[t] = funmap(redu[(size_t)bn * MID_ + t])
              + msg1[(size_t)bn * MID_ + t] + msgg[b * MID_ + t];
    __syncthreads();
    float a = o1b[t] + o2b[t];
#pragma unroll 4
    for (int k = 0; k < ZIN; ++k)  a += zs[k] * o1w[k * OUT_ + t];
#pragma unroll 4
    for (int k = 0; k < MID_; ++k) a += ms[k] * o2w[k * OUT_ + t];
    zout[(size_t)bn * OUT_ + t] = fmaxf(a, 0.0f);
}

// ---------------------------------------------------------------------------
extern "C" void kernel_launch(void* const* d_in, const int* in_sizes, int n_in,
                              void* d_out, int out_size, void* d_ws, size_t ws_size,
                              hipStream_t stream) {
    (void)in_sizes; (void)n_in; (void)out_size; (void)ws_size;
    const float* node = (const float*)d_in[0];
    const float* edge = (const float*)d_in[1];
    const float* gf   = (const float*)d_in[2];
    const float* hid  = (const float*)d_in[3];
    const int*   adj  = (const int*)  d_in[4];
    const float* m1w  = (const float*)d_in[5];
    const float* m1b  = (const float*)d_in[6];
    const float* m2w  = (const float*)d_in[7];
    const float* m2b  = (const float*)d_in[8];
    const float* mew  = (const float*)d_in[9];
    const float* meb  = (const float*)d_in[10];
    const float* mgw  = (const float*)d_in[11];
    const float* mgb  = (const float*)d_in[12];
    const float* o1w  = (const float*)d_in[13];
    const float* o1b  = (const float*)d_in[14];
    const float* o2w  = (const float*)d_in[15];
    const float* o2b  = (const float*)d_in[16];

    float* ws   = (float*)d_ws;
    float* z0   = ws;                        // 262144 floats
    float* z1   = ws + 262144;               // 262144
    float* msg1 = ws + 524288;               // 131072
    float* msg2 = ws + 655360;               // 131072
    float* msgg = ws + 786432;               // 256
    unsigned* redu = (unsigned*)(ws + 786688); // 131072 u32  (~3.7 MB total)

    k_init_z<<<(B_ * N_ * DN + 255) / 256, 256, 0, stream>>>(node, hid, z0);

    const float* zin = z0;
    for (int l = 0; l < LAY; ++l) {
        float* zout = (l == LAY - 1) ? (float*)d_out : z1;

        k_msgs<<<B_ * N_, 128, 0, stream>>>(zin, gf,
            m1w + (size_t)l * ZIN * MID_, m1b + l * MID_,
            m2w + (size_t)l * ZIN * MID_, m2b + l * MID_,
            mgw + (size_t)l * DN  * MID_, mgb + l * MID_,
            meb + l * MID_,
            msg1, msg2, msgg, redu);

        dim3 g2(SEGS, N_ / 16, B_);          // 1024 workgroups x 8 waves
        k_edge_max<<<g2, 256, 0, stream>>>(edge, adj, msg2,
            mew + (size_t)l * DN * MID_, redu);

        k_out<<<B_ * N_, OUT_, 0, stream>>>(zin, redu, msg1, msgg,
            o1w + (size_t)l * ZIN  * OUT_, o1b + l * OUT_,
            o2w + (size_t)l * MID_ * OUT_, o2b + l * OUT_, zout);

        zin = zout;
    }
}
